// GatingNetwork_4707284156656
// MI455X (gfx1250) — matrix operand order repaired and checked by reading, not compiled, not run on hardware
//
#include <hip/hip_runtime.h>

typedef __attribute__((ext_vector_type(16))) __bf16 v16bf;
typedef __attribute__((ext_vector_type(8)))  __bf16 v8bf;
typedef __attribute__((ext_vector_type(8)))  float  v8f;
typedef __attribute__((ext_vector_type(4)))  float  v4f;

constexpr int NROWS = 16384;
constexpr int DIM   = 4096;
constexpr int NEXP  = 128;
constexpr int TOPK  = 8;
constexpr int BM    = 64;    // rows per block = 4 waves * 16
constexpr int KSTEP = 64;    // K per pipeline stage (two WMMA K-slices)
constexpr int LSTR  = 72;    // W LDS row stride, bf16 (64 + 8 pad, b128 conflict-free)
constexpr int XSTR  = 68;    // x LDS row stride, f32  (64 + 4 pad, b128 conflict-free)
constexpr int NSTEPS = DIM / KSTEP;

// ---- kernel 0: one-shot W fp32 -> bf16 (W is 2MB, L2-resident) ----
__global__ __launch_bounds__(256)
void wconv(const float* __restrict__ Wf, __bf16* __restrict__ Wb) {
    int i = (blockIdx.x * 256 + threadIdx.x) * 8;
    v4f a = *(const v4f*)(Wf + i);
    v4f c = *(const v4f*)(Wf + i + 4);
    v8bf o;
    #pragma unroll
    for (int j = 0; j < 4; ++j) { o[j] = (__bf16)a[j]; o[4 + j] = (__bf16)c[j]; }
    *(v8bf*)(Wb + i) = o;
}

template<bool ASYNCW>
__global__ __launch_bounds__(128)
void gating_topk_softmax(const float* __restrict__ x,
                         const float* __restrict__ Wf,
                         const __bf16* __restrict__ Wb,
                         const float* __restrict__ bias,
                         float* __restrict__ out)
{
    __shared__ float  xsb[3][BM * XSTR];     // triple-buffered x tile (fp32), 51KB
    __shared__ __bf16 wsb[3][NEXP * LSTR];   // triple-buffered W tile (bf16), 54KB

    const int tid  = threadIdx.x;
    const int wave = tid >> 5;
    const int lane = tid & 31;
    const int lh   = lane >> 4;      // half-wave: 0 or 1
    const int col  = lane & 15;
    const int blockBase = blockIdx.x * BM;
    const int rowTile   = blockBase + wave * 16;

    v8f acc[8];
    #pragma unroll
    for (int t = 0; t < 8; ++t)
        #pragma unroll
        for (int j = 0; j < 8; ++j) acc[t][j] = 0.0f;

    // ---- async staging: x tile (64x64 f32) + W tile (128x64 bf16); 16 asyncs/thread ----
    auto stageAsync = [&](int buf, int k0) {
        #pragma unroll
        for (int i = 0; i < 8; ++i) {           // x: 8 x 16B chunks per thread
            int idx = tid + i * 128;
            int row = idx >> 4, g = idx & 15;
            unsigned int ldsoff =
                (unsigned int)(uintptr_t)(&xsb[buf][row * XSTR + g * 4]);
            unsigned long long gaddr =
                (unsigned long long)(uintptr_t)(x + (long)(blockBase + row) * DIM + k0 + g * 4);
            asm volatile("global_load_async_to_lds_b128 %0, %1, off"
                         :: "v"(ldsoff), "v"(gaddr) : "memory");
        }
        #pragma unroll
        for (int i = 0; i < 8; ++i) {           // W: 8 x 16B chunks per thread
            int idx = tid + i * 128;
            int e = idx >> 3, g = idx & 7;
            unsigned int ldsoff =
                (unsigned int)(uintptr_t)(&wsb[buf][e * LSTR + g * 8]);
            unsigned long long gaddr =
                (unsigned long long)(uintptr_t)(Wb + (long)e * DIM + k0 + g * 8);
            asm volatile("global_load_async_to_lds_b128 %0, %1, off"
                         :: "v"(ldsoff), "v"(gaddr) : "memory");
        }
    };
    auto stageCvt = [&](int buf, int k0) {   // fallback: synchronous reg staging
        #pragma unroll
        for (int i = 0; i < 8; ++i) {           // x copy
            int idx = tid + i * 128;
            int row = idx >> 4, g = idx & 15;
            v4f a = *(const v4f*)(x + (long)(blockBase + row) * DIM + k0 + g * 4);
            *(v4f*)&xsb[buf][row * XSTR + g * 4] = a;
        }
        #pragma unroll
        for (int i = 0; i < 8; ++i) {           // W fp32 -> bf16
            int idx = tid + i * 128;
            int e = idx >> 3, g = idx & 7;
            const float* p = Wf + (long)e * DIM + k0 + g * 8;
            v4f a = *(const v4f*)p;
            v4f c = *(const v4f*)(p + 4);
            v8bf o;
            #pragma unroll
            for (int j = 0; j < 4; ++j) { o[j] = (__bf16)a[j]; o[4 + j] = (__bf16)c[j]; }
            *(v8bf*)&wsb[buf][e * LSTR + g * 8] = o;
        }
    };

    // A slice from LDS: lanes 0-15 hold K {0..7,16..23}, lanes 16-31 {8..15,24..31}
    auto ldA = [&](int buf, int kk, v4f* r) {
        const float* ap = &xsb[buf][(wave * 16 + col) * XSTR + kk + lh * 8];
        r[0] = *(const v4f*)ap;
        r[1] = *(const v4f*)(ap + 4);
        r[2] = *(const v4f*)(ap + 16);
        r[3] = *(const v4f*)(ap + 20);
    };
    auto cvtA = [&](const v4f* r) {
        v16bf a;
        #pragma unroll
        for (int j = 0; j < 4; ++j) {
            a[j]      = (__bf16)r[0][j];
            a[4 + j]  = (__bf16)r[1][j];
            a[8 + j]  = (__bf16)r[2][j];
            a[12 + j] = (__bf16)r[3][j];
        }
        return a;
    };
    // B 32x16: lanes 0-15 hold K {0..7,16..23}, lanes 16-31 {8..15,24..31}; N = lane%16
    auto ldB = [&](int buf, int tt, v8bf* slot) {
        int t = tt & 7;
        int kk = (tt >> 3) * 32;
        const __bf16* bp = &wsb[buf][(t * 16 + col) * LSTR + kk + lh * 8];
        slot[0] = *(const v8bf*)bp;
        slot[1] = *(const v8bf*)(bp + 16);
    };

    // 16 WMMAs per stage; everything reads LDS with pipelined counted waits
    auto computeK64 = [&](int buf) {
        v4f ar[4];
        ldA(buf, 0, ar);                 // 4 ds loads, issued first
        v8bf pb[4][2];
        ldB(buf, 0, pb[0]);
        ldB(buf, 1, pb[1]);
        ldB(buf, 2, pb[2]);
        v16bf a0 = cvtA(ar);
        v16bf a1;
        v4f ar2[4];
        #pragma unroll
        for (int tt = 0; tt < 16; ++tt) {
            if (tt == 2) ldA(buf, 32, ar2);        // prefetch slice 1 (LDS)
            if (tt == 5) a1 = cvtA(ar2);           // convert while a0 still live
            if (tt < 13) ldB(buf, tt + 3, pb[(tt + 3) & 3]);
            v16bf bb;
            #pragma unroll
            for (int j = 0; j < 8; ++j) {
                bb[j]     = pb[tt & 3][0][j];
                bb[8 + j] = pb[tt & 3][1][j];
            }
            acc[tt & 7] = __builtin_amdgcn_wmma_f32_16x16x32_bf16(
                false, (tt < 8) ? a0 : a1, false, bb, (short)0, acc[tt & 7],
                false, false);
        }
    };

    // ---- main loop: triple-buffered stages, async prefetch distance 2 ----
    if constexpr (ASYNCW) {
        stageAsync(0, 0);
        stageAsync(1, KSTEP);
        asm volatile("s_wait_asynccnt 0x10" ::: "memory");   // buffer 0 complete
    } else {
        stageCvt(0, 0);
        stageCvt(1, KSTEP);
    }
    __syncthreads();

    for (int s = 0; s < NSTEPS; ++s) {
        const bool issue = (s + 2 < NSTEPS);
        if (issue) {
            if constexpr (ASYNCW) stageAsync((s + 2) % 3, (s + 2) * KSTEP);
            else stageCvt((s + 2) % 3, (s + 2) * KSTEP);
        }
        computeK64(s % 3);
        if constexpr (ASYNCW) {
            if (s + 1 < NSTEPS) {
                if (issue) asm volatile("s_wait_asynccnt 0x10" ::: "memory");
                else       asm volatile("s_wait_asynccnt 0x0"  ::: "memory");
            }
        }
        __syncthreads();
    }

    // ---- epilogue: bias + top-8 threshold + masked softmax (all in-wave) ----
    // acc[t][j] = logit(row rowTile + lh*8 + j, expert t*16 + col)
    float bv[8];
    #pragma unroll
    for (int t = 0; t < 8; ++t) bv[t] = bias[t * 16 + col];

    const float NEG = -3.0e38f;
    #pragma unroll
    for (int j = 0; j < 8; ++j) {
        float v[8], tmp[8];
        #pragma unroll
        for (int t = 0; t < 8; ++t) { v[t] = acc[t][j] + bv[t]; tmp[t] = v[t]; }

        float rmax = NEG, th = NEG;
        #pragma unroll
        for (int it = 0; it < TOPK; ++it) {
            float lm = tmp[0];
            #pragma unroll
            for (int t = 1; t < 8; ++t) lm = fmaxf(lm, tmp[t]);
            #pragma unroll
            for (int o = 1; o < 16; o <<= 1)        // xor<16 stays inside half-wave
                lm = fmaxf(lm, __shfl_xor(lm, o, 32));
            if (it == 0) rmax = lm;
            th = lm;
            #pragma unroll
            for (int t = 0; t < 8; ++t) if (tmp[t] == lm) tmp[t] = NEG;
        }

        float e[8], ssum = 0.0f;
        #pragma unroll
        for (int t = 0; t < 8; ++t) {
            float ev = (v[t] >= th) ? __expf(v[t] - rmax) : 0.0f;
            e[t] = ev;
            ssum += ev;
        }
        #pragma unroll
        for (int o = 1; o < 16; o <<= 1) ssum += __shfl_xor(ssum, o, 32);
        float inv = 1.0f / ssum;

        long r = rowTile + lh * 8 + j;
        #pragma unroll
        for (int t = 0; t < 8; ++t)
            out[r * NEXP + t * 16 + col] = e[t] * inv;
    }
}

extern "C" void kernel_launch(void* const* d_in, const int* in_sizes, int n_in,
                              void* d_out, int out_size, void* d_ws, size_t ws_size,
                              hipStream_t stream) {
    (void)in_sizes; (void)n_in; (void)out_size;
    const float* x = (const float*)d_in[0];
    const float* W = (const float*)d_in[1];
    const float* b = (const float*)d_in[2];
    // d_in[3] is k == 8 (hardcoded as TOPK)
    float* out = (float*)d_out;

    const size_t wbBytes = (size_t)NEXP * DIM * sizeof(__bf16);  // 1 MB
    if (ws_size >= wbBytes) {
        __bf16* Wb = (__bf16*)d_ws;
        wconv<<<dim3(NEXP * DIM / 8 / 256), dim3(256), 0, stream>>>(W, Wb);
        gating_topk_softmax<true><<<dim3(NROWS / BM), dim3(128), 0, stream>>>(
            x, W, Wb, b, out);
    } else {
        gating_topk_softmax<false><<<dim3(NROWS / BM), dim3(128), 0, stream>>>(
            x, W, nullptr, b, out);
    }
}
